// FPModule_26834955666010
// MI455X (gfx1250) — matrix operand order repaired and checked by reading, not compiled, not run on hardware
//
#include <hip/hip_runtime.h>
#include <stdint.h>

typedef __attribute__((ext_vector_type(16))) __bf16 v16bf;
typedef __attribute__((ext_vector_type(8)))  float  v8f;

#define B_DIM  8
#define N_PTS  2048
#define M_PTS  8192
#define C_IN   256
#define CS_IN  128
#define H_DIM  512
#define ROWS   (B_DIM * M_PTS)   /* 65536 */
#define K1     (C_IN + CS_IN)    /* 384   */

// ---- gfx1250 async global->LDS copy (ASYNCcnt path), guarded so absence can't regress
#if __has_builtin(__builtin_amdgcn_global_load_async_to_lds_b128) && \
    __has_builtin(__builtin_amdgcn_s_wait_asynccnt)
#define HAVE_ASYNC_LDS 1
typedef int v4i_vs __attribute__((vector_size(16)));
typedef __attribute__((address_space(1))) v4i_vs gbl_v4i;   // prints as "int4 __device__*"
typedef __attribute__((address_space(3))) v4i_vs lds_v4i;   // prints as "int4 __shared__*"
__device__ __forceinline__ void async_copy_b128(const void* gaddr, void* ldsaddr) {
  __builtin_amdgcn_global_load_async_to_lds_b128(
      (gbl_v4i*)(uintptr_t)gaddr,
      (lds_v4i*)(uint32_t)(uintptr_t)ldsaddr,   // AS3 is 32-bit; low bits = LDS offset
      0, 0);
}
#else
#define HAVE_ASYNC_LDS 0
#endif

__device__ __forceinline__ uint16_t f2bf(float f) {
  union { float f; uint32_t u; } a; a.f = f;
  uint32_t u = a.u;
  u += 0x7FFFu + ((u >> 16) & 1u);     // round-to-nearest-even
  return (uint16_t)(u >> 16);
}

// ---------------------------------------------------------------- weights->bf16
__global__ __launch_bounds__(256) void cvt_f32_bf16(const float* __restrict__ in,
                                                    uint16_t* __restrict__ out, int n) {
  int i = blockIdx.x * 256 + threadIdx.x;
  if (i < n) out[i] = f2bf(in[i]);
}

// ---------------------------------------------------------------- kNN (K=3)
__global__ __launch_bounds__(256) void knn_kernel(const float* __restrict__ pos,
                                                  const float* __restrict__ pos_skip,
                                                  int* __restrict__ idx_out,
                                                  float* __restrict__ w_out) {
  __shared__ float ps[256 * 3];
  const int tid = threadIdx.x;
  const int q   = blockIdx.x * 256 + tid;      // query row (blocks never straddle a batch)
  const int b   = q / M_PTS;
  const int base = b * N_PTS;

  const float qx = pos_skip[q * 3 + 0];
  const float qy = pos_skip[q * 3 + 1];
  const float qz = pos_skip[q * 3 + 2];

  float d0 = 3.4e38f, d1 = 3.4e38f, d2v = 3.4e38f;
  int   i0 = base, i1 = base, i2 = base;

  for (int nb = 0; nb < N_PTS; nb += 256) {
    __syncthreads();
    const int sr = (base + nb + tid) * 3;
    ps[tid * 3 + 0] = pos[sr + 0];
    ps[tid * 3 + 1] = pos[sr + 1];
    ps[tid * 3 + 2] = pos[sr + 2];
    __syncthreads();
#pragma unroll 4
    for (int j = 0; j < 256; ++j) {
      const float dx = qx - ps[j * 3 + 0];
      const float dy = qy - ps[j * 3 + 1];
      const float dz = qz - ps[j * 3 + 2];
      const float d  = dx * dx + dy * dy + dz * dz;
      if (d < d2v) {
        const int n = base + nb + j;
        if (d < d0)      { d2v = d1; i2 = i1; d1 = d0; i1 = i0; d0 = d; i0 = n; }
        else if (d < d1) { d2v = d1; i2 = i1; d1 = d;  i1 = n; }
        else             { d2v = d;  i2 = n; }
      }
    }
  }
  const float w0 = 1.0f / fmaxf(d0,  1e-16f);
  const float w1 = 1.0f / fmaxf(d1,  1e-16f);
  const float w2 = 1.0f / fmaxf(d2v, 1e-16f);
  const float inv = 1.0f / (w0 + w1 + w2);
  idx_out[q * 3 + 0] = i0;  idx_out[q * 3 + 1] = i1;  idx_out[q * 3 + 2] = i2;
  w_out[q * 3 + 0] = w0 * inv;  w_out[q * 3 + 1] = w1 * inv;  w_out[q * 3 + 2] = w2 * inv;
}

// ------------------------------------------- interpolate + concat -> bf16 h0 (ROWS x 384)
__global__ __launch_bounds__(256) void interp_kernel(const float* __restrict__ x,
                                                     const float* __restrict__ x_skip,
                                                     const int* __restrict__ idx,
                                                     const float* __restrict__ wn,
                                                     uint16_t* __restrict__ h0) {
  const int q = blockIdx.x;
  const int c = threadIdx.x;
  const int i0 = idx[q * 3 + 0], i1 = idx[q * 3 + 1], i2 = idx[q * 3 + 2];
  const float w0 = wn[q * 3 + 0], w1 = wn[q * 3 + 1], w2 = wn[q * 3 + 2];
  const float v = w0 * x[(size_t)i0 * C_IN + c] +
                  w1 * x[(size_t)i1 * C_IN + c] +
                  w2 * x[(size_t)i2 * C_IN + c];
  h0[(size_t)q * K1 + c] = f2bf(v);
  if (c < CS_IN)
    h0[(size_t)q * K1 + C_IN + c] = f2bf(x_skip[(size_t)q * CS_IN + c]);
}

// ---------------------------------------------------------------- WMMA GEMM + bias + relu
// C[M,N] = relu(A[M,K] * B[K,N] + bias), A/B bf16, f32 accumulate.
// Block: 256 thr = 8 wave32, tile 128x128; wave(wr 0..3, wc 0..1) -> 32x64 = 2x4 wmma tiles.
template <bool OUT_BF16>
__global__ __launch_bounds__(256) void gemm_bias_relu(const uint16_t* __restrict__ A,
                                                      const uint16_t* __restrict__ Bw,
                                                      const float* __restrict__ bias,
                                                      void* __restrict__ outp,
                                                      int K, int N) {
  constexpr int LDA_S = 40;   // 128x32 A tile, padded stride (banks)
  constexpr int LDB_S = 136;  // 32x128 B tile, padded stride
  __shared__ uint16_t As[128 * LDA_S];
  __shared__ uint16_t Bs[32 * LDB_S];

  const int tid  = threadIdx.x;
  const int lane = tid & 31;
  const int w    = tid >> 5;
  const int wr   = w & 3;
  const int wc   = w >> 2;
  const int m0   = blockIdx.y * 128;
  const int n0   = blockIdx.x * 128;

  v8f acc[2][4];
#pragma unroll
  for (int mt = 0; mt < 2; ++mt)
#pragma unroll
    for (int nt = 0; nt < 4; ++nt)
#pragma unroll
      for (int r = 0; r < 8; ++r) acc[mt][nt][r] = 0.0f;

  // cooperative staging pointers: A -> 2 threads/row x 16 elems; B -> 8 threads/row x 16 elems
  const int arow_l = tid >> 1, akoff = (tid & 1) * 16;
  const int brow_l = tid >> 3, bcoff = (tid & 7) * 16;
  const uint16_t* aG = A + (size_t)(m0 + arow_l) * K + akoff;
  const uint16_t* bG = Bw + (size_t)brow_l * N + n0 + bcoff;
  uint16_t* aS = &As[arow_l * LDA_S + akoff];
  uint16_t* bS = &Bs[brow_l * LDB_S + bcoff];

  for (int k0 = 0; k0 < K; k0 += 32) {
#if HAVE_ASYNC_LDS
    // gfx1250 async copy: global -> LDS without VGPR round trip (ASYNCcnt)
    async_copy_b128(aG,     aS);
    async_copy_b128(aG + 8, aS + 8);
    async_copy_b128(bG,     bS);
    async_copy_b128(bG + 8, bS + 8);
#else
    *(uint4*)(aS)     = *(const uint4*)(aG);
    *(uint4*)(aS + 8) = *(const uint4*)(aG + 8);
    *(uint4*)(bS)     = *(const uint4*)(bG);
    *(uint4*)(bS + 8) = *(const uint4*)(bG + 8);
#endif
    aG += 32;
    bG += (size_t)32 * N;
    if (k0 + 32 < K) {                 // gfx1250 global_prefetch_b8 of next tiles
      __builtin_prefetch(aG, 0, 1);
      __builtin_prefetch(bG, 0, 1);
    }
#if HAVE_ASYNC_LDS
    __builtin_amdgcn_s_wait_asynccnt(0);   // this wave's tile copies landed in LDS
#endif
    __syncthreads();

    // A fragment (ISA 16-bit A 16x32): lanes 0-15 row m, K 0-7 & 16-23; lanes 16-31 K 8-15 & 24-31
    union { v16bf v; uint4 u[2]; } af[2], bfr[4];
    const int klo = (lane < 16) ? 0 : 8;
    const int ar  = wr * 32 + (lane & 15);
#pragma unroll
    for (int mt = 0; mt < 2; ++mt) {
      const uint16_t* p = &As[(ar + mt * 16) * LDA_S + klo];
      af[mt].u[0] = *(const uint4*)(p);
      af[mt].u[1] = *(const uint4*)(p + 16);
    }
    // B fragment: lane l holds column N = l%16, K same order as A (klo..klo+7, klo+16..klo+23)
#pragma unroll
    for (int nt = 0; nt < 4; ++nt) {
      const uint16_t* p = &Bs[klo * LDB_S + wc * 64 + nt * 16 + (lane & 15)];
      bfr[nt].u[0] = make_uint4((uint32_t)p[0 * LDB_S] | ((uint32_t)p[1 * LDB_S] << 16),
                                (uint32_t)p[2 * LDB_S] | ((uint32_t)p[3 * LDB_S] << 16),
                                (uint32_t)p[4 * LDB_S] | ((uint32_t)p[5 * LDB_S] << 16),
                                (uint32_t)p[6 * LDB_S] | ((uint32_t)p[7 * LDB_S] << 16));
      bfr[nt].u[1] = make_uint4((uint32_t)p[16 * LDB_S] | ((uint32_t)p[17 * LDB_S] << 16),
                                (uint32_t)p[18 * LDB_S] | ((uint32_t)p[19 * LDB_S] << 16),
                                (uint32_t)p[20 * LDB_S] | ((uint32_t)p[21 * LDB_S] << 16),
                                (uint32_t)p[22 * LDB_S] | ((uint32_t)p[23 * LDB_S] << 16));
    }
#pragma unroll
    for (int mt = 0; mt < 2; ++mt)
#pragma unroll
      for (int nt = 0; nt < 4; ++nt)
        acc[mt][nt] = __builtin_amdgcn_wmma_f32_16x16x32_bf16(
            false, af[mt].v, false, bfr[nt].v, (short)0, acc[mt][nt], false, false);
    __syncthreads();
  }

  // epilogue: C/D layout — VGPR r, lanes 0-15: M=r, N=lane; lanes 16-31: M=r+8, N=lane-16
#pragma unroll
  for (int mt = 0; mt < 2; ++mt) {
#pragma unroll
    for (int nt = 0; nt < 4; ++nt) {
      const int col = n0 + wc * 64 + nt * 16 + (lane & 15);
      const int rb  = m0 + wr * 32 + mt * 16 + ((lane >> 4) << 3);
      const float bv = bias[col];
#pragma unroll
      for (int r = 0; r < 8; ++r) {
        const float v = fmaxf(acc[mt][nt][r] + bv, 0.0f);
        const size_t o = (size_t)(rb + r) * N + col;
        if (OUT_BF16) ((uint16_t*)outp)[o] = f2bf(v);
        else          ((float*)outp)[o] = v;
      }
    }
  }
}

// ---------------------------------------------------------------- host launcher
extern "C" void kernel_launch(void* const* d_in, const int* in_sizes, int n_in,
                              void* d_out, int out_size, void* d_ws, size_t ws_size,
                              hipStream_t stream) {
  const float* x        = (const float*)d_in[0];
  const float* pos      = (const float*)d_in[1];
  const float* x_skip   = (const float*)d_in[3];
  const float* pos_skip = (const float*)d_in[4];
  const float* W1       = (const float*)d_in[6];
  const float* b1       = (const float*)d_in[7];
  const float* W2       = (const float*)d_in[8];
  const float* b2       = (const float*)d_in[9];

  char* ws = (char*)d_ws;
  int*      idx = (int*)     (ws + 0);          // 65536*3*4   = 786432
  float*    wn  = (float*)   (ws + 786432);     // 65536*3*4   = 786432
  uint16_t* w1b = (uint16_t*)(ws + 1572864);    // 384*512*2   = 393216
  uint16_t* w2b = (uint16_t*)(ws + 1966080);    // 512*512*2   = 524288
  uint16_t* h0  = (uint16_t*)(ws + 2490368);    // 65536*384*2 = 50331648
  uint16_t* h1  = (uint16_t*)(ws + 52822016);   // 65536*512*2 = 67108864

  cvt_f32_bf16<<<(K1 * H_DIM + 255) / 256, 256, 0, stream>>>(W1, w1b, K1 * H_DIM);
  cvt_f32_bf16<<<(H_DIM * H_DIM + 255) / 256, 256, 0, stream>>>(W2, w2b, H_DIM * H_DIM);

  knn_kernel<<<ROWS / 256, 256, 0, stream>>>(pos, pos_skip, idx, wn);
  interp_kernel<<<ROWS, 256, 0, stream>>>(x, x_skip, idx, wn, h0);

  dim3 g(H_DIM / 128, ROWS / 128);  // (4, 512)
  gemm_bias_relu<true ><<<g, 256, 0, stream>>>(h0, w1b, b1, (void*)h1, K1,    H_DIM);
  gemm_bias_relu<false><<<g, 256, 0, stream>>>(h1, w2b, b2, d_out,     H_DIM, H_DIM);
}
